// NeuralSymbolicIntegrator_38457137168829
// MI455X (gfx1250) — matrix operand order repaired and checked by reading, not compiled, not run on hardware
//
#include <hip/hip_runtime.h>
#include <hip/hip_bf16.h>

typedef __attribute__((ext_vector_type(16))) _Float16 v16h;
typedef __attribute__((ext_vector_type(8)))  _Float16 v8h;
typedef __attribute__((ext_vector_type(4)))  _Float16 v4h;
typedef __attribute__((ext_vector_type(8)))  float    v8f;

#define EPS_F 1e-8f
#define THRESH_F 0.75f

union AFrag { v16h v; v8h h[2]; };

// ---------------------------------------------------------------------------
// f32 -> f16 elementwise convert (n is a multiple of 4 for all our tensors)
// ---------------------------------------------------------------------------
__global__ __launch_bounds__(256) void cvt_f32_f16_kernel(
    const float* __restrict__ in, _Float16* __restrict__ out, int n) {
  int i = (blockIdx.x * 256 + threadIdx.x) * 4;
  if (i >= n) return;
  float4 v = *(const float4*)(in + i);
  v4h o;
  o[0] = (_Float16)v.x; o[1] = (_Float16)v.y;
  o[2] = (_Float16)v.z; o[3] = (_Float16)v.w;
  *(v4h*)(out + i) = o;
}

// ---------------------------------------------------------------------------
// Row normalize (cols == 512), one wave32 per row, f32 in -> normalized f16 out
// scale = 1 / max(||row||, eps)   (matches torch cosine_similarity eps)
// ---------------------------------------------------------------------------
__global__ __launch_bounds__(256) void normalize_rows_512_kernel(
    const float* __restrict__ in, _Float16* __restrict__ out, int rows) {
  int row  = blockIdx.x * 8 + (threadIdx.x >> 5);
  int lane = threadIdx.x & 31;
  if (row >= rows) return;

  const float* p = in + (size_t)row * 512 + lane * 16;
  float4 v[4];
  float ss = 0.f;
#pragma unroll
  for (int i = 0; i < 4; ++i) {
    v[i] = *(const float4*)(p + i * 4);
    ss += v[i].x * v[i].x + v[i].y * v[i].y + v[i].z * v[i].z + v[i].w * v[i].w;
  }
#pragma unroll
  for (int off = 16; off > 0; off >>= 1) ss += __shfl_xor(ss, off, 32);
  float scale = 1.f / fmaxf(sqrtf(ss), EPS_F);

  _Float16* q = out + (size_t)row * 512 + lane * 16;
  v8h o0, o1;
  o0[0] = (_Float16)(v[0].x * scale); o0[1] = (_Float16)(v[0].y * scale);
  o0[2] = (_Float16)(v[0].z * scale); o0[3] = (_Float16)(v[0].w * scale);
  o0[4] = (_Float16)(v[1].x * scale); o0[5] = (_Float16)(v[1].y * scale);
  o0[6] = (_Float16)(v[1].z * scale); o0[7] = (_Float16)(v[1].w * scale);
  o1[0] = (_Float16)(v[2].x * scale); o1[1] = (_Float16)(v[2].y * scale);
  o1[2] = (_Float16)(v[2].z * scale); o1[3] = (_Float16)(v[2].w * scale);
  o1[4] = (_Float16)(v[3].x * scale); o1[5] = (_Float16)(v[3].y * scale);
  o1[6] = (_Float16)(v[3].z * scale); o1[7] = (_Float16)(v[3].w * scale);
  *(v8h*)q       = o0;
  *(v8h*)(q + 8) = o1;
}

// ---------------------------------------------------------------------------
// WMMA GEMM:  C[M,N] = A[M,K] @ Bt[N,K]^T   (both f16 row-major, f32 accum)
// Block: 256 thr = 8 waves; block tile 128x128; wave tile 64x32 (4x2 16x16).
// Requires: K % 32 == 0, N % 32 == 0, M % 128 == 0 (true for all 3 GEMMs).
// EPI: 0 = bias+relu -> f16 out | 1 = bias -> f32 out | 2 = threshold -> f32
// ---------------------------------------------------------------------------
template <int EPI>
__global__ __launch_bounds__(256) void wmma_gemm_kernel(
    const _Float16* __restrict__ A, const _Float16* __restrict__ Bt,
    const float* __restrict__ bias, void* __restrict__ outp,
    int M, int N, int K) {
  const int tid  = threadIdx.x;
  const int wave = tid >> 5;
  const int lane = tid & 31;
  const int wm   = wave >> 2;      // 0..1
  const int wn   = wave & 3;       // 0..3
  const int m0   = blockIdx.y * 128 + wm * 64;
  const int n0   = blockIdx.x * 128 + wn * 32;
  if (n0 >= N) return;             // whole-wave guard (EXEC stays all-ones)

  const int ml   = lane & 15;
  const int kioA = (lane & 16) ? 16 : 0;  // A: hi-half lanes take K+16
  const int kbB  = (lane & 16) ? 16 : 0;  // B: hi-half lanes take K+16

  const _Float16* arow[4];
  const _Float16* brow[2];
#pragma unroll
  for (int i = 0; i < 4; ++i)
    arow[i] = A + (size_t)(m0 + i * 16 + ml) * K + kioA;
#pragma unroll
  for (int j = 0; j < 2; ++j)
    brow[j] = Bt + (size_t)(n0 + j * 16 + ml) * K + kbB;

  v8f acc[4][2] = {};

  for (int kb = 0; kb < K; kb += 32) {
    if (kb + 32 < K) {                     // gfx1250 global_prefetch_b8 path
      __builtin_prefetch(brow[0] + kb + 32, 0, 3);
      __builtin_prefetch(brow[1] + kb + 32, 0, 3);
    }
    AFrag a[4];
    v16h  b[2];
#pragma unroll
    for (int i = 0; i < 4; ++i) {          // two b128 loads per A fragment
      a[i].h[0] = *(const v8h*)(arow[i] + kb);
      a[i].h[1] = *(const v8h*)(arow[i] + kb + 8);
    }
#pragma unroll
    for (int j = 0; j < 2; ++j)            // contiguous 32B per B fragment
      b[j] = *(const v16h*)(brow[j] + kb);

#pragma unroll
    for (int i = 0; i < 4; ++i)
#pragma unroll
      for (int j = 0; j < 2; ++j)
        acc[i][j] = __builtin_amdgcn_wmma_f32_16x16x32_f16(
            false, a[i].v, false, b[j], (short)0, acc[i][j], false, false);
  }

  // C/D layout: VGPR r, lane l -> M = r + (l<16?0:8), N = l%16
  const int rbase = (lane & 16) ? 8 : 0;
  const int col   = lane & 15;
#pragma unroll
  for (int i = 0; i < 4; ++i) {
#pragma unroll
    for (int j = 0; j < 2; ++j) {
      const int n = n0 + j * 16 + col;
      float bv = 0.f;
      if constexpr (EPI == 0 || EPI == 1) bv = bias[n];
#pragma unroll
      for (int r = 0; r < 8; ++r) {
        const int m = m0 + i * 16 + rbase + r;
        float s = acc[i][j][r] + bv;
        if constexpr (EPI == 0) {
          ((_Float16*)outp)[(size_t)m * N + n] = (_Float16)fmaxf(s, 0.f);
        } else if constexpr (EPI == 1) {
          ((float*)outp)[(size_t)m * N + n] = s;
        } else {
          ((float*)outp)[(size_t)m * N + n] = (s > THRESH_F) ? s : 0.f;
        }
      }
    }
  }
}

// ---------------------------------------------------------------------------
extern "C" void kernel_launch(void* const* d_in, const int* in_sizes, int n_in,
                              void* d_out, int out_size, void* d_ws, size_t ws_size,
                              hipStream_t stream) {
  const float* X  = (const float*)d_in[0];   // [B, Din]
  const float* W1 = (const float*)d_in[1];   // [Dh, Din]
  const float* b1 = (const float*)d_in[2];   // [Dh]
  const float* W2 = (const float*)d_in[3];   // [Dout, Dh]
  const float* b2 = (const float*)d_in[4];   // [Dout]
  const float* C  = (const float*)d_in[5];   // [N, Dout]

  const int Dh   = in_sizes[2];
  const int Dout = in_sizes[4];
  const int Din  = in_sizes[1] / Dh;
  const int B    = in_sizes[0] / Din;
  const int N    = in_sizes[5] / Dout;

  // Workspace carve-out (256B-aligned slices)
  char*  ws  = (char*)d_ws;
  size_t off = 0;
  auto carve = [&](size_t bytes) -> char* {
    char* p = ws + off;
    off += (bytes + 255) & ~(size_t)255;
    return p;
  };
  _Float16* Ch  = (_Float16*)carve((size_t)N * Dout * 2);   // normalized concepts f16
  _Float16* Xh  = (_Float16*)carve((size_t)B * Din * 2);
  _Float16* W1h = (_Float16*)carve((size_t)Dh * Din * 2);
  _Float16* W2h = (_Float16*)carve((size_t)Dout * Dh * 2);
  _Float16* Hh  = (_Float16*)carve((size_t)B * Dh * 2);     // relu hidden f16
  float*    Yf  = (float*)   carve((size_t)B * Dout * 4);   // neural_output f32
  _Float16* Qh  = (_Float16*)carve((size_t)B * Dout * 2);   // normalized query f16

  const dim3 blk(256);

  // 1) f32 -> f16 converts
  int nX = B * Din, nW1 = Dh * Din, nW2 = Dout * Dh;
  cvt_f32_f16_kernel<<<(nX  / 4 + 255) / 256, blk, 0, stream>>>(X,  Xh,  nX);
  cvt_f32_f16_kernel<<<(nW1 / 4 + 255) / 256, blk, 0, stream>>>(W1, W1h, nW1);
  cvt_f32_f16_kernel<<<(nW2 / 4 + 255) / 256, blk, 0, stream>>>(W2, W2h, nW2);

  // 2) h = relu(X @ W1^T + b1)  -> f16
  dim3 g1((Dh + 127) / 128, (B + 127) / 128);
  wmma_gemm_kernel<0><<<g1, blk, 0, stream>>>(Xh, W1h, b1, Hh, B, Dh, Din);

  // 3) y = h @ W2^T + b2  -> f32
  dim3 g2((Dout + 127) / 128, (B + 127) / 128);
  wmma_gemm_kernel<1><<<g2, blk, 0, stream>>>(Hh, W2h, b2, Yf, B, Dout, Dh);

  // 4) normalize rows -> f16 (query and concepts; both have 512 cols)
  normalize_rows_512_kernel<<<(B + 7) / 8, blk, 0, stream>>>(Yf, Qh, B);
  normalize_rows_512_kernel<<<(N + 7) / 8, blk, 0, stream>>>(C,  Ch, N);

  // 5) sims = qn @ cn^T, thresholded -> f32 output [B, N]
  dim3 g3((N + 127) / 128, (B + 127) / 128);
  wmma_gemm_kernel<2><<<g3, blk, 0, stream>>>(Qh, Ch, nullptr, d_out, B, N, Dout);
}
